// NativeBitLinear_47141561041541
// MI455X (gfx1250) — matrix operand order repaired and checked by reading, not compiled, not run on hardware
//
#include <hip/hip_runtime.h>

// ---------------------------------------------------------------------------
// BitLinear: out(32x11008) = x(32x4096) @ (sign * group_scale).T
// Memory-bound (180 MB sign stream @ 23.3 TB/s ~ 7.8 us floor).
// WMMA f16 16x16x32 keeps math off the critical path.
// ---------------------------------------------------------------------------

typedef __attribute__((ext_vector_type(16))) _Float16 v16h;
typedef __attribute__((ext_vector_type(8)))  float    v8f;
typedef __attribute__((ext_vector_type(4)))  int      v4i;
typedef __attribute__((ext_vector_type(4)))  float    v4f;
typedef __attribute__((ext_vector_type(2)))  __fp16   fp16x2;

#define OUT_FEATURES 11008
#define IN_FEATURES  4096
#define BATCH        32
#define GROUPS_PER_ROW (IN_FEATURES / 128)   // 32
#define SPLIT_K      8
#define K_CHUNK      (IN_FEATURES / SPLIT_K) // 512
#define K_STEPS      (K_CHUNK / 32)          // 16 wmma K-steps per chunk
#define WAVES_PER_WG 8
#define WG_THREADS   (WAVES_PER_WG * 32)
#define N_WGS        (OUT_FEATURES / (WAVES_PER_WG * 16)) // 86
#define LDS_BYTES    (K_STEPS * 64 * 32)     // 32 KB of A fragments

// pack two f32 -> packed f16 pair (v_cvt_pk_rtz_f16_f32)
static __device__ __forceinline__ unsigned int pk_f16(float a, float b) {
    union { fp16x2 h; unsigned int u; } v;
    v.h = __builtin_amdgcn_cvt_pkrtz(a, b);
    return v.u;
}

// two int32 signs (+1 / -1) -> packed f16 (+1.0 / -1.0).
// sign byte of a ±1 int32 is 0x00 / 0xFF; v_perm_b32 gathers the two sign
// bytes into bytes 1 and 3 (bytes 0,2 forced to 0x00 via SEL=0x0C), then a
// single and+or (v_and_or/v_bfi) builds 0x3C00 | signbit per half.
static __device__ __forceinline__ unsigned int pk_sign(unsigned int u0, unsigned int u1) {
    // perm operands: (hi=u1 -> byte idx 4..7, lo=u0 -> byte idx 0..3)
    // dst byte0 = 0x00 (sel 0x0C), byte1 = u0.byte3 (sel 3),
    // dst byte2 = 0x00 (sel 0x0C), byte3 = u1.byte3 (sel 7)
    unsigned int t = __builtin_amdgcn_perm(u1, u0, 0x070C030Cu);
    return (t & 0x80008000u) | 0x3C003C00u;
}

__global__ void zero_f32_kernel(float* __restrict__ p, int n) {
    int i = blockIdx.x * blockDim.x + threadIdx.x;
    if (i < n) p[i] = 0.0f;
}

__global__ __launch_bounds__(WG_THREADS)
void bitlinear_wmma_kernel(const float* __restrict__ x,
                           const float* __restrict__ scales,
                           const int*   __restrict__ signs,
                           float*       __restrict__ out) {
    extern __shared__ unsigned int lds_u32[];

    const int tid  = threadIdx.x;
    const int lane = tid & 31;
    const int wave = tid >> 5;
    const int k0   = blockIdx.y * K_CHUNK;

    // ---- Stage x chunk into LDS as pre-swizzled f16 WMMA A-fragments ----
    // Fragment f = (kstep s, m-tile, lane l): 8 dwords = 16 halves.
    // 16-bit A layout: lanes 0-15 hold K {0..7,16..23}, lanes 16-31 hold
    // K {8..15,24..31} of row M = m*16 + (l&15).
    for (int f = tid; f < K_STEPS * 64; f += WG_THREADS) {
        const int s   = f >> 6;
        const int rem = f & 63;
        const int m   = rem >> 5;
        const int l   = rem & 31;
        const int M   = m * 16 + (l & 15);
        const int run0 = k0 + s * 32 + ((l & 16) ? 8 : 0);
        const float* xp = x + M * IN_FEATURES + run0;
        v4f a0 = *reinterpret_cast<const v4f*>(xp);
        v4f a1 = *reinterpret_cast<const v4f*>(xp + 4);
        v4f b0 = *reinterpret_cast<const v4f*>(xp + 16);
        v4f b1 = *reinterpret_cast<const v4f*>(xp + 20);
        unsigned int* dst = lds_u32 + f * 8;
        dst[0] = pk_f16(a0.x, a0.y); dst[1] = pk_f16(a0.z, a0.w);
        dst[2] = pk_f16(a1.x, a1.y); dst[3] = pk_f16(a1.z, a1.w);
        dst[4] = pk_f16(b0.x, b0.y); dst[5] = pk_f16(b0.z, b0.w);
        dst[6] = pk_f16(b1.x, b1.y); dst[7] = pk_f16(b1.z, b1.w);
    }
    __syncthreads();

    // ---- Per-wave output tile: 16 out-features x 32 batch rows ----
    const int ncol  = (blockIdx.x * WAVES_PER_WG + wave) * 16 + (lane & 15);
    const int khalf = lane >> 4;           // B layout: K {0..7,16..23} vs {8..15,24..31}
    const int rowbase = ncol * IN_FEATURES;

    v8f acc0 = {};
    v8f acc1 = {};

    for (int g = 0; g < K_CHUNK / 128; ++g) {   // 4 scale-groups per chunk
        v8f t0 = {};
        v8f t1 = {};
        if (g < (K_CHUNK / 128) - 1) {
            __builtin_prefetch(signs + rowbase + k0 + (g + 1) * 128 + khalf * 64, 0, 0);
        }
#pragma unroll
        for (int s2 = 0; s2 < 4; ++s2) {        // 4 K-steps of 32 per group
            const int kk = g * 4 + s2;
            // B fragment: lane holds K {8h..8h+7, 16+8h..16+8h+7} of weight row ncol
            const v4i* sp = reinterpret_cast<const v4i*>(
                signs + rowbase + k0 + kk * 32 + khalf * 8);
            v4i w0 = __builtin_nontemporal_load(sp);
            v4i w1 = __builtin_nontemporal_load(sp + 1);
            v4i w2 = __builtin_nontemporal_load(sp + 4);
            v4i w3 = __builtin_nontemporal_load(sp + 5);
            union { unsigned int u[8]; v16h h; } bf;
            bf.u[0] = pk_sign(w0.x, w0.y); bf.u[1] = pk_sign(w0.z, w0.w);
            bf.u[2] = pk_sign(w1.x, w1.y); bf.u[3] = pk_sign(w1.z, w1.w);
            bf.u[4] = pk_sign(w2.x, w2.y); bf.u[5] = pk_sign(w2.z, w2.w);
            bf.u[6] = pk_sign(w3.x, w3.y); bf.u[7] = pk_sign(w3.z, w3.w);

            // A fragments from LDS (pre-swizzled)
            union { v4i i2[2]; v16h h; } a0u, a1u;
            const v4i* l0 = reinterpret_cast<const v4i*>(
                lds_u32 + (kk * 64 + 0 * 32 + lane) * 8);
            const v4i* l1 = reinterpret_cast<const v4i*>(
                lds_u32 + (kk * 64 + 1 * 32 + lane) * 8);
            a0u.i2[0] = l0[0]; a0u.i2[1] = l0[1];
            a1u.i2[0] = l1[0]; a1u.i2[1] = l1[1];

            t0 = __builtin_amdgcn_wmma_f32_16x16x32_f16(
                false, a0u.h, false, bf.h, (short)0, t0, false, false);
            t1 = __builtin_amdgcn_wmma_f32_16x16x32_f16(
                false, a1u.h, false, bf.h, (short)0, t1, false, false);
        }
        // apply per-(row, group) scale outside the WMMA chain;
        // every accumulator element in this lane belongs to column ncol.
        const float sc = scales[ncol * GROUPS_PER_ROW + (int)blockIdx.y * 4 + g];
#pragma unroll
        for (int j = 0; j < 8; ++j) {
            acc0[j] += sc * t0[j];
            acc1[j] += sc * t1[j];
        }
    }

    // ---- Split-K reduction into out (C/D layout: lane -> column ncol,
    //      VGPR j -> batch row j + khalf*8 (+16 for second M tile)) ----
    const int mrow = khalf * 8;
#pragma unroll
    for (int j = 0; j < 8; ++j) {
        atomicAdd(out + (mrow + j) * OUT_FEATURES + ncol, acc0[j]);
        atomicAdd(out + (16 + mrow + j) * OUT_FEATURES + ncol, acc1[j]);
    }
}

extern "C" void kernel_launch(void* const* d_in, const int* in_sizes, int n_in,
                              void* d_out, int out_size, void* d_ws, size_t ws_size,
                              hipStream_t stream) {
    const float* x      = (const float*)d_in[0];
    const float* scales = (const float*)d_in[1];
    const int*   signs  = (const int*)d_in[2];
    float*       out    = (float*)d_out;

    // zero output (split-K accumulates with f32 atomics)
    zero_f32_kernel<<<(out_size + 255) / 256, 256, 0, stream>>>(out, out_size);

    dim3 grid(N_WGS, SPLIT_K);
    dim3 block(WG_THREADS);
    bitlinear_wmma_kernel<<<grid, block, LDS_BYTES, stream>>>(x, scales, signs, out);
}
